// SelfAttention_33389075759655
// MI455X (gfx1250) — matrix operand order repaired and checked by reading, not compiled, not run on hardware
//
#include <hip/hip_runtime.h>
#include <hip/hip_bf16.h>
#include <math.h>

// Self-attention fused pipeline for MI455X (gfx1250, wave32, WMMA).
// ws layout: Q(33.5MB) K(33.5MB) Vt(33.5MB) seq_f16(2MB) W_f16(1.5MB)
// => needs ws_size >= ~104.4 MB.

typedef __attribute__((ext_vector_type(16))) _Float16 v16h;
typedef __attribute__((ext_vector_type(8)))  _Float16 v8h;
typedef __attribute__((ext_vector_type(8)))  float    v8f;

#define NB   8
#define NS   1024
#define NE   128
#define NH   16
#define HE   (NH * NE)       // 2048
#define F3   (3 * HE)        // 6144
#define NM   (NB * NS)       // 8192 rows
#define QK_SCALE 0.08838834764831845f   // 1/sqrt(128)

// ---------------------------------------------------------------------------
// Kernel 0: fp32 -> f16 conversion (8 elements / thread, vectorized)
// ---------------------------------------------------------------------------
__global__ __launch_bounds__(256) void cvt_kernel(const float* __restrict__ src,
                                                  _Float16* __restrict__ dst) {
    const size_t i = ((size_t)blockIdx.x * blockDim.x + threadIdx.x) * 8;
    const float4 a = *(const float4*)(src + i);
    const float4 b = *(const float4*)(src + i + 4);
    v8h o;
    o[0]=(_Float16)a.x; o[1]=(_Float16)a.y; o[2]=(_Float16)a.z; o[3]=(_Float16)a.w;
    o[4]=(_Float16)b.x; o[5]=(_Float16)b.y; o[6]=(_Float16)b.z; o[7]=(_Float16)b.w;
    *(v8h*)(dst + i) = o;
}

static __device__ __forceinline__ v16h a_operand_f16(const _Float16* p, int g) {
    // A-operand (16x32 f16): halves 0..15 = K(g*16..g*16+15)
    v8h lo = *(const v8h*)(p + g * 16);
    v8h hi = *(const v8h*)(p + g * 16 + 8);
    v16h a;
#pragma unroll
    for (int i = 0; i < 8; ++i) { a[i] = lo[i]; a[8 + i] = hi[i]; }
    return a;
}

// ---------------------------------------------------------------------------
// Kernel 1: QKV projection GEMM + bias + fused RoPE.
// One wave per 16(M) x 128(F) tile = one head's full feature range, so the
// RoPE rotation partners (e, e+64) sit in the same lane (c[nt], c[nt+4]).
// Q pre-scaled by 1/sqrt(E).  Q/K -> [B,H,S,E] f16, V -> transposed [B,H,E,S].
// ---------------------------------------------------------------------------
__global__ __launch_bounds__(32) void qkv_kernel(const _Float16* __restrict__ seqh,
                                                 const _Float16* __restrict__ Wh,
                                                 const float* __restrict__ bias,
                                                 _Float16* __restrict__ Qb,
                                                 _Float16* __restrict__ Kb,
                                                 _Float16* __restrict__ Vt) {
    const int lane = threadIdx.x & 31;
    const int lm = lane & 15;
    const int g  = lane >> 4;
    const int row0 = blockIdx.y * 16;        // M tile (m = b*S + s)
    const int f0   = blockIdx.x * 128;       // feature tile = one (which, head)
    const int which = f0 >> 11;              // 0=Q 1=K 2=V (uniform per block)
    const int h     = (f0 >> 7) & (NH - 1);

    // A tile: 16 rows x 128 K (f16), four K=32 chunks, resident
    v16h a[4];
    const _Float16* arow = seqh + (size_t)(row0 + lm) * NE;
#pragma unroll
    for (int kk = 0; kk < 4; ++kk) a[kk] = a_operand_f16(arow + kk * 32, g);

    // 8 accumulator tiles (16x16 each) covering e = 0..127
    v8f c[8];
#pragma unroll
    for (int nt = 0; nt < 8; ++nt) c[nt] = (v8f){};

#pragma unroll
    for (int nt = 0; nt < 8; ++nt) {
        const _Float16* brow = Wh + (size_t)(f0 + nt * 16 + lm) * NE;
#pragma unroll
        for (int kk = 0; kk < 4; ++kk) {
            v16h bv = *(const v16h*)(brow + kk * 32 + g * 16);
            c[nt] = __builtin_amdgcn_wmma_f32_16x16x32_f16(false, a[kk], false, bv,
                                                           (short)0, c[nt], false, false);
        }
    }

    float bb[8];
#pragma unroll
    for (int nt = 0; nt < 8; ++nt) bb[nt] = bias[f0 + nt * 16 + lm];

    if (which == 2) {
        // V: no RoPE; scatter into transposed [B,H,E,S] layout
#pragma unroll
        for (int nt = 0; nt < 8; ++nt) {
            const int e = nt * 16 + lm;
#pragma unroll
            for (int r = 0; r < 8; ++r) {
                const int m = row0 + r + 8 * g;
                const int b = m >> 10;
                const int s = m & (NS - 1);
                Vt[(((size_t)(b * NH + h) * NE + e) * NS) + s] =
                    (_Float16)(c[nt][r] + bb[nt]);
            }
        }
    } else {
        _Float16* dst = (which == 0) ? Qb : Kb;
        const float scale = (which == 0) ? QK_SCALE : 1.0f;
#pragma unroll
        for (int nt = 0; nt < 4; ++nt) {
            const int j = nt * 16 + lm;                       // rotation index 0..63
            const float invf = __powf(10000.0f, -(float)j * (1.0f / 64.0f));
#pragma unroll
            for (int r = 0; r < 8; ++r) {
                const int m = row0 + r + 8 * g;
                const int b = m >> 10;
                const int s = m & (NS - 1);
                float sn, cs;
                __sincosf((float)s * invf, &sn, &cs);
                const float lo = c[nt][r]     + bb[nt];
                const float hi = c[nt + 4][r] + bb[nt + 4];
                const size_t base = ((size_t)(b * NH + h) * NS + s) * NE;
                dst[base + j]      = (_Float16)((lo * cs - hi * sn) * scale);
                dst[base + j + 64] = (_Float16)((lo * sn + hi * cs) * scale);
            }
        }
    }
}

// ---------------------------------------------------------------------------
// Kernel 2: flash attention.  One wave per (b,h, 16-query tile); streams keys
// in chunks of 32 with online softmax; P reshaped C->A layout via LDS.
// Output written as (B, S, H, E) fp32.
// ---------------------------------------------------------------------------
__global__ __launch_bounds__(32) void fa_kernel(const _Float16* __restrict__ Qb,
                                                const _Float16* __restrict__ Kb,
                                                const _Float16* __restrict__ Vt,
                                                float* __restrict__ out) {
    const int lane = threadIdx.x & 31;
    const int lm = lane & 15;
    const int g  = lane >> 4;
    const int bid = blockIdx.x;
    const int qt = bid & 63;          // 1024/16 query tiles
    const int bh = bid >> 6;          // 0..127
    const int h  = bh & (NH - 1);
    const int b  = bh >> 4;
    const int q0 = qt * 16;

    __shared__ _Float16 pl[16][48];   // probs staging; 96B row stride (16B-aligned)

    // Resident Q tile (A operands, already RoPE'd and scaled)
    v16h qa[4];
    const _Float16* qrow = Qb + ((size_t)bh * NS + q0 + lm) * NE;
#pragma unroll
    for (int kk = 0; kk < 4; ++kk) qa[kk] = a_operand_f16(qrow + kk * 32, g);

    v8f o[8];
#pragma unroll
    for (int t = 0; t < 8; ++t) o[t] = (v8f){};
    float mrun[8], lrun[8];
#pragma unroll
    for (int r = 0; r < 8; ++r) { mrun[r] = -INFINITY; lrun[r] = 0.0f; }

    const _Float16* kbase = Kb + (size_t)bh * NS * NE;
    const _Float16* vbase = Vt + (size_t)bh * NE * NS;

    for (int k0 = 0; k0 < NS; k0 += 32) {
        if (k0 + 32 < NS) {   // uniform branch: prefetch next key chunk
            __builtin_prefetch(kbase + (size_t)(k0 + 32 + lm) * NE, 0, 1);
            __builtin_prefetch(kbase + (size_t)(k0 + 48 + lm) * NE, 0, 1);
        }

        // ---- scores: 16q x 32k = 2 WMMA C tiles, 4 K-chunks over d ----
        v8f c0 = {}, c1 = {};
        const _Float16* krow0 = kbase + (size_t)(k0 + lm) * NE;
        const _Float16* krow1 = kbase + (size_t)(k0 + 16 + lm) * NE;
#pragma unroll
        for (int kk = 0; kk < 4; ++kk) {
            v16h b0 = *(const v16h*)(krow0 + kk * 32 + g * 16);
            v16h b1 = *(const v16h*)(krow1 + kk * 32 + g * 16);
            c0 = __builtin_amdgcn_wmma_f32_16x16x32_f16(false, qa[kk], false, b0,
                                                        (short)0, c0, false, false);
            c1 = __builtin_amdgcn_wmma_f32_16x16x32_f16(false, qa[kk], false, b1,
                                                        (short)0, c1, false, false);
        }

        // ---- online softmax (C-layout rows: r + 8*g, cols across 16 lanes) ----
#pragma unroll
        for (int r = 0; r < 8; ++r) {
            float mx = fmaxf(c0[r], c1[r]);
            mx = fmaxf(mx, __shfl_xor(mx, 1, 32));
            mx = fmaxf(mx, __shfl_xor(mx, 2, 32));
            mx = fmaxf(mx, __shfl_xor(mx, 4, 32));
            mx = fmaxf(mx, __shfl_xor(mx, 8, 32));
            const float mnew  = fmaxf(mrun[r], mx);
            const float alpha = __expf(mrun[r] - mnew);
            const float p0 = __expf(c0[r] - mnew);
            const float p1 = __expf(c1[r] - mnew);
            float rs = p0 + p1;
            rs += __shfl_xor(rs, 1, 32);
            rs += __shfl_xor(rs, 2, 32);
            rs += __shfl_xor(rs, 4, 32);
            rs += __shfl_xor(rs, 8, 32);
            lrun[r] = lrun[r] * alpha + rs;
            mrun[r] = mnew;
#pragma unroll
            for (int t = 0; t < 8; ++t) o[t][r] *= alpha;
            pl[r + 8 * g][lm]      = (_Float16)p0;
            pl[r + 8 * g][16 + lm] = (_Float16)p1;
        }
        __builtin_amdgcn_wave_barrier();

        // ---- P (C-layout) -> A-operand layout via LDS (16B vector loads) ----
        v8h plo = *(const v8h*)(&pl[lm][g * 16]);
        v8h phi = *(const v8h*)(&pl[lm][g * 16 + 8]);
        v16h pa;
#pragma unroll
        for (int i = 0; i < 8; ++i) { pa[i] = plo[i]; pa[8 + i] = phi[i]; }
        __builtin_amdgcn_wave_barrier();

        // ---- O += P * V  (V^T rows contiguous over keys) ----
#pragma unroll
        for (int t = 0; t < 8; ++t) {
            const _Float16* vrow = vbase + (size_t)(t * 16 + lm) * NS + k0;
            v16h vb = *(const v16h*)(vrow + g * 16);
            o[t] = __builtin_amdgcn_wmma_f32_16x16x32_f16(false, pa, false, vb,
                                                          (short)0, o[t], false, false);
        }
    }

    // ---- normalize + store (B, S, H, E) ----
    float inv[8];
#pragma unroll
    for (int r = 0; r < 8; ++r) inv[r] = 1.0f / lrun[r];
#pragma unroll
    for (int t = 0; t < 8; ++t) {
#pragma unroll
        for (int r = 0; r < 8; ++r) {
            const int s = q0 + r + 8 * g;
            const int e = t * 16 + lm;
            out[(((size_t)b * NS + s) * NH + h) * NE + e] = o[t][r] * inv[r];
        }
    }
}

// ---------------------------------------------------------------------------
extern "C" void kernel_launch(void* const* d_in, const int* in_sizes, int n_in,
                              void* d_out, int out_size, void* d_ws, size_t ws_size,
                              hipStream_t stream) {
    const float* seq  = (const float*)d_in[0];
    const float* W    = (const float*)d_in[1];
    const float* bias = (const float*)d_in[2];
    float* out = (float*)d_out;

    const size_t nQ   = (size_t)NB * NH * NS * NE;  // 16,777,216 halfs per tensor
    const size_t nSeq = (size_t)NM * NE;            // 1,048,576
    const size_t nW   = (size_t)F3 * NE;            //   786,432
    _Float16* Qb   = (_Float16*)d_ws;
    _Float16* Kb   = Qb + nQ;
    _Float16* Vt   = Kb + nQ;
    _Float16* seqh = Vt + nQ;
    _Float16* Wh   = seqh + nSeq;

    // 0) one-time fp32 -> f16 conversion of GEMM operands
    cvt_kernel<<<(int)(nSeq / 8 / 256), 256, 0, stream>>>(seq, seqh);
    cvt_kernel<<<(int)(nW   / 8 / 256), 256, 0, stream>>>(W, Wh);

    // 1) QKV projection + bias + fused RoPE: tiles of 16 (M) x 128 (F)
    dim3 g1(F3 / 128, NM / 16);                     // (48, 512)
    qkv_kernel<<<g1, 32, 0, stream>>>(seqh, Wh, bias, Qb, Kb, Vt);

    // 2) Flash attention: one wave per (b, h, 16-query tile)
    fa_kernel<<<NB * NH * (NS / 16), 32, 0, stream>>>(Qb, Kb, Vt, out);
}
